// IntegratedDCALLM_45071386804652
// MI455X (gfx1250) — matrix operand order repaired and checked by reading, not compiled, not run on hardware
//
#include <hip/hip_runtime.h>
#include <cstddef>
#include <cstdint>

// ---------------------------------------------------------------------------
// Model constants (from the reference)
// ---------------------------------------------------------------------------
#define D_MODEL 1024
#define SEQ     2048
#define WIN     256
#define NWIN    8
#define NHEAD   16
#define NBATCH  2
#define HD      64
#define NLAYER  4
#define DFF     4096
#define NVOCAB  50257

typedef unsigned short ushort_t;

// ---------------------------------------------------------------------------
// CDNA5 WMMA types (wave32, 16x16x32 bf16 -> f32)
// ---------------------------------------------------------------------------
typedef __attribute__((ext_vector_type(16))) __bf16       v16bf;
typedef __attribute__((ext_vector_type(8)))  float        v8f;
typedef __attribute__((ext_vector_type(4)))  float        f32x4;
typedef __attribute__((ext_vector_type(4)))  unsigned int u32x4;
typedef __attribute__((ext_vector_type(2)))  unsigned int u32x2;

union BF16Frag { v16bf v; u32x4 q[2]; };

__device__ inline unsigned int f2bfu(float f) {
  unsigned int u = __float_as_uint(f);
  u += 0x7FFFu + ((u >> 16) & 1u);   // RNE f32 -> bf16 bits
  return u >> 16;
}

__device__ inline v8f vzero8() {
  v8f z;
#pragma unroll
  for (int i = 0; i < 8; ++i) z[i] = 0.f;
  return z;
}

__device__ inline v8f wmma_bf16(const BF16Frag& a, const BF16Frag& b, v8f c) {
  return __builtin_amdgcn_wmma_f32_16x16x32_bf16(false, a.v, false, b.v,
                                                 (short)0, c, false, false);
}

__device__ inline float gelu_f(float x) {
  return 0.5f * x * (1.f + tanhf(0.7978845608028654f * (x + 0.044715f * x * x * x)));
}

// ---------------------------------------------------------------------------
// CDNA5 async copy: 16B global -> LDS, tracked by ASYNCcnt
// LDS byte address = low 32 bits of generic pointer (flat aperture rule)
// ---------------------------------------------------------------------------
__device__ inline void async_b128(unsigned lds_byte_off, const void* gaddr) {
  asm volatile("global_load_async_to_lds_b128 %0, %1, off"
               :: "v"(lds_byte_off), "v"(gaddr) : "memory");
}
__device__ inline void wait_async0() {
  asm volatile("s_wait_asynccnt 0x0" ::: "memory");
}
__device__ inline unsigned lds_off_of(const void* p) {
  return (unsigned)(unsigned long long)(uintptr_t)p;
}

// ---------------------------------------------------------------------------
// Weight conversion: f32 -> bf16, optionally transposed  Wt[N][K] = W[K][N]
// ---------------------------------------------------------------------------
__global__ void __launch_bounds__(256) cvt_kernel(const float* __restrict__ in,
                                                  ushort_t* __restrict__ out,
                                                  int n4) {
  const int i = blockIdx.x * 256 + threadIdx.x;
  if (i < n4) {
    f32x4 f = *(const f32x4*)(in + (size_t)i * 4);
    u32x2 pk;
    pk.x = f2bfu(f.x) | (f2bfu(f.y) << 16);
    pk.y = f2bfu(f.z) | (f2bfu(f.w) << 16);
    *(u32x2*)(out + (size_t)i * 4) = pk;
  }
}

__global__ void __launch_bounds__(256) cvt_t_kernel(const float* __restrict__ W,
                                                    ushort_t* __restrict__ Wt,
                                                    int K, int N) {
  __shared__ float tile[32][33];
  const size_t mat = (size_t)blockIdx.z * (size_t)K * N;
  const int n0 = blockIdx.x * 32, k0 = blockIdx.y * 32;
  const int tx = threadIdx.x & 31, ty = threadIdx.x >> 5;   // 32 x 8
#pragma unroll
  for (int j = 0; j < 4; ++j)
    tile[ty + j * 8][tx] = W[mat + (size_t)(k0 + ty + j * 8) * N + n0 + tx];
  __syncthreads();
#pragma unroll
  for (int j = 0; j < 4; ++j)
    Wt[mat + (size_t)(n0 + ty + j * 8) * K + k0 + tx] =
        (ushort_t)f2bfu(tile[tx][ty + j * 8]);
}

// ---------------------------------------------------------------------------
// Embedding (f32 residual stream): x = tok_emb[id] + pos_emb[s]
// ---------------------------------------------------------------------------
__global__ void __launch_bounds__(256) embed_kernel(const int* __restrict__ ids,
                                                    const float* __restrict__ tok,
                                                    const float* __restrict__ pos,
                                                    float* __restrict__ X) {
  const int row = blockIdx.x;
  const int s   = row & (SEQ - 1);
  const int id  = ids[row];
  const int c   = threadIdx.x * 4;
  f32x4 tv = *(const f32x4*)(tok + (size_t)id * D_MODEL + c);
  f32x4 pv = *(const f32x4*)(pos + (size_t)s * D_MODEL + c);
  f32x4 o;
  o.x = tv.x + pv.x; o.y = tv.y + pv.y; o.z = tv.z + pv.z; o.w = tv.w + pv.w;
  *(f32x4*)(X + (size_t)row * D_MODEL + c) = o;
}

// ---------------------------------------------------------------------------
// Compressed past (bf16 out): comp[b,t,:] = mean_{r<4} emb[b,4t+r,:]
// ---------------------------------------------------------------------------
__global__ void __launch_bounds__(256) compress_kernel(const float* __restrict__ Emb,
                                                       ushort_t* __restrict__ Comp) {
  const int t  = blockIdx.x;
  const int b  = t >> 9;
  const int tt = t & 511;
  const int c  = threadIdx.x * 4;
  const float* base = Emb + ((size_t)(b * SEQ + tt * 4)) * D_MODEL + c;
  f32x4 a0 = *(const f32x4*)(base + 0 * D_MODEL);
  f32x4 a1 = *(const f32x4*)(base + 1 * D_MODEL);
  f32x4 a2 = *(const f32x4*)(base + 2 * D_MODEL);
  f32x4 a3 = *(const f32x4*)(base + 3 * D_MODEL);
  u32x2 pk;
  pk.x = f2bfu(0.25f * (a0.x + a1.x + a2.x + a3.x)) |
         (f2bfu(0.25f * (a0.y + a1.y + a2.y + a3.y)) << 16);
  pk.y = f2bfu(0.25f * (a0.z + a1.z + a2.z + a3.z)) |
         (f2bfu(0.25f * (a0.w + a1.w + a2.w + a3.w)) << 16);
  *(u32x2*)(Comp + (size_t)t * D_MODEL + c) = pk;
}

// ---------------------------------------------------------------------------
// LayerNorm (f32 in, bf16 out — output only ever feeds GEMM A operands)
// ---------------------------------------------------------------------------
__global__ void __launch_bounds__(256) ln_kernel(const float* __restrict__ X,
                                                 const float* __restrict__ g,
                                                 const float* __restrict__ be,
                                                 ushort_t* __restrict__ Y) {
  __shared__ float rs[8], rq[8];
  __shared__ float mu_s, rstd_s;
  const int row = blockIdx.x;
  const int c   = threadIdx.x * 4;
  f32x4 xv = *(const f32x4*)(X + (size_t)row * D_MODEL + c);
  float s = xv.x + xv.y + xv.z + xv.w;
  float q = xv.x * xv.x + xv.y * xv.y + xv.z * xv.z + xv.w * xv.w;
#pragma unroll
  for (int d = 1; d < 32; d <<= 1) {
    s += __shfl_xor(s, d, 32);
    q += __shfl_xor(q, d, 32);
  }
  const int wave = threadIdx.x >> 5, lane = threadIdx.x & 31;
  if (lane == 0) { rs[wave] = s; rq[wave] = q; }
  __syncthreads();
  if (threadIdx.x == 0) {
    float ts = 0.f, tq = 0.f;
#pragma unroll
    for (int i = 0; i < 8; ++i) { ts += rs[i]; tq += rq[i]; }
    const float mu  = ts * (1.f / (float)D_MODEL);
    const float var = tq * (1.f / (float)D_MODEL) - mu * mu;
    mu_s = mu;
    rstd_s = rsqrtf(var + 1e-5f);
  }
  __syncthreads();
  const float mu = mu_s, r = rstd_s;
  f32x4 gv = *(const f32x4*)(g + c);
  f32x4 bv = *(const f32x4*)(be + c);
  u32x2 pk;
  pk.x = f2bfu((xv.x - mu) * r * gv.x + bv.x) |
         (f2bfu((xv.y - mu) * r * gv.y + bv.y) << 16);
  pk.y = f2bfu((xv.z - mu) * r * gv.z + bv.z) |
         (f2bfu((xv.w - mu) * r * gv.w + bv.w) << 16);
  *(u32x2*)(Y + (size_t)row * D_MODEL + c) = pk;
}

// ---------------------------------------------------------------------------
// WMMA GEMM (bf16 x bf16 -> f32 accum)
//   A  : bf16 [M][lda]          (activations)
//   Bt : bf16 [N][ldb]          (pre-transposed weight, ldb == K)
//   out: Cf (f32) and/or Cb (bf16), epilogue bias/gelu/residual/skip-win0
// Double-buffered K tiles staged with global_load_async_to_lds_b128.
// Block 256 (8 waves); tile 128x128x32; 8 v_wmma per wave per K step.
// ---------------------------------------------------------------------------
#define BM 128
#define BN 128
#define BK 32
#define ASTR 40          // LDS row stride (bf16) -> conflict-free b128 frags
#define TILE_US (BM * ASTR)   // 5120 ushorts per tile buffer

__global__ void __launch_bounds__(256) gemm_bf16(
    const ushort_t* __restrict__ A, const ushort_t* __restrict__ Bt,
    const float* __restrict__ bias, const float* __restrict__ Res,
    float* __restrict__ Cf, ushort_t* __restrict__ Cb,
    int N, int K, int lda, int ldb, int ldc,
    int do_gelu, int add_res, int skip_w0) {
  __shared__ ushort_t sm[4 * TILE_US];   // [buf][A|B] : A0 B0 A1 B1

  const int t    = threadIdx.x;
  const int wave = t >> 5, lane = t & 31;
  const int lg   = lane >> 4, ln = lane & 15;
  const int m0   = blockIdx.y * BM;
  const int n0   = blockIdx.x * BN;
  const int mbase = (wave >> 2) * 64;
  const int nbase = (wave & 3) * 32;
  const unsigned lds0 = lds_off_of(sm);

  v8f acc[4][2];
#pragma unroll
  for (int i = 0; i < 4; ++i)
#pragma unroll
    for (int j = 0; j < 2; ++j) acc[i][j] = vzero8();

  const int nk = K / BK;

  // async-stage one K tile (A + B panels) into buffer `buf`
  auto stage = [&](int kt, int buf) {
    const int k0 = kt * BK;
    const unsigned ab = lds0 + (unsigned)(buf * 2 * TILE_US) * 2;
    const unsigned bb = ab + (unsigned)TILE_US * 2;
#pragma unroll
    for (int i = 0; i < 2; ++i) {
      const int idx = t + i * 256;        // 0..511 : 16B chunks of A tile
      const int row = idx >> 2;
      const int c   = (idx & 3) << 3;     // 0,8,16,24 (bf16 units)
      async_b128(ab + (unsigned)(row * ASTR + c) * 2,
                 A + (size_t)(m0 + row) * lda + k0 + c);
    }
#pragma unroll
    for (int i = 0; i < 2; ++i) {
      const int idx = t + i * 256;        // B panel rows are output columns
      const int row = idx >> 2;
      const int c   = (idx & 3) << 3;
      async_b128(bb + (unsigned)(row * ASTR + c) * 2,
                 Bt + (size_t)(n0 + row) * ldb + k0 + c);
    }
  };

  stage(0, 0);
  wait_async0();
  __syncthreads();

  for (int kt = 0; kt < nk; ++kt) {
    const int cur = kt & 1;
    if (kt + 1 < nk) stage(kt + 1, cur ^ 1);

    const ushort_t* As = sm + cur * 2 * TILE_US;
    const ushort_t* Bs = As + TILE_US;

    BF16Frag af[4], bf[2];
#pragma unroll
    for (int mt = 0; mt < 4; ++mt) {
      const int row = mbase + mt * 16 + ln;
      af[mt].q[0] = *(const u32x4*)(&As[row * ASTR + lg * 8]);
      af[mt].q[1] = *(const u32x4*)(&As[row * ASTR + lg * 8 + 16]);
    }
#pragma unroll
    for (int nt = 0; nt < 2; ++nt) {
      const int col = nbase + nt * 16 + ln;
      bf[nt].q[0] = *(const u32x4*)(&Bs[col * ASTR + lg * 8]);
      bf[nt].q[1] = *(const u32x4*)(&Bs[col * ASTR + lg * 8 + 16]);
    }
#pragma unroll
    for (int mt = 0; mt < 4; ++mt)
#pragma unroll
      for (int nt = 0; nt < 2; ++nt)
        acc[mt][nt] = wmma_bf16(af[mt], bf[nt], acc[mt][nt]);

    if (kt + 1 < nk) wait_async0();
    __syncthreads();
  }

  // epilogue (C layout: lane ln = N, vgpr v + 8*lg = M)
#pragma unroll
  for (int mt = 0; mt < 4; ++mt) {
#pragma unroll
    for (int nt = 0; nt < 2; ++nt) {
      const int gn = n0 + nbase + nt * 16 + ln;
      if (gn >= N) continue;
#pragma unroll
      for (int v = 0; v < 8; ++v) {
        const int gm = m0 + mbase + mt * 16 + v + 8 * lg;
        float val = acc[mt][nt][v];
        if (bias) val += bias[gn];
        if (do_gelu) val = gelu_f(val);
        const size_t off = (size_t)gm * ldc + gn;
        if (add_res) {
          const float rv = Res[off];
          val = (skip_w0 && ((gm & (SEQ - 1)) < WIN)) ? rv : (val + rv);
        }
        if (Cf) Cf[off] = val;
        if (Cb) Cb[off] = (ushort_t)f2bfu(val);
      }
    }
  }
}

// ---------------------------------------------------------------------------
// Flash attention over one (batch, head, window); bf16 in, bf16 out.
// Q/K panels async-staged to LDS; V staged transposed; online softmax in
// the WMMA C layout; P -> A-fragment via per-wave LDS round trip.
// ---------------------------------------------------------------------------
__global__ void __launch_bounds__(256) attn_kernel(
    const ushort_t* __restrict__ Qp, const ushort_t* __restrict__ Kp,
    const ushort_t* __restrict__ Vp, ushort_t* __restrict__ Ob,
    int q_bstride, int k_bstride, int k_winmul,
    int kvlen_base, int kvlen_mul, int wi0, int causal, int kvcap) {
  extern __shared__ ushort_t smem_us[];
  const int wi    = blockIdx.x + wi0;
  const int hh    = blockIdx.y;
  const int bb    = blockIdx.z;
  const int qrow0 = bb * q_bstride + wi * WIN;
  const int krow0 = bb * k_bstride + wi * k_winmul;
  const int kvlen = kvlen_base + wi * kvlen_mul;
  const int col0  = hh * HD;
  const int vstride = kvcap + 8;

  ushort_t* Qs = smem_us;            // [256][72]
  ushort_t* Ks = Qs + 256 * 72;      // [kvcap][72]
  ushort_t* Vs = Ks + kvcap * 72;    // [64][vstride] (transposed V)
  ushort_t* Ps = Vs + 64 * vstride;  // per-wave [2][16][40]
  const unsigned qs0 = lds_off_of(Qs);
  const unsigned ks0 = lds_off_of(Ks);

  const int t = threadIdx.x;
  // ---- async stage Q (256 x 64 bf16)
#pragma unroll
  for (int i = 0; i < 8; ++i) {
    const int idx = t + i * 256;        // 2048 x 16B chunks
    const int row = idx >> 3;
    const int c   = (idx & 7) << 3;
    async_b128(qs0 + (unsigned)(row * 72 + c) * 2,
               Qp + (size_t)(qrow0 + row) * D_MODEL + col0 + c);
  }
  // ---- async stage K (kvlen x 64 bf16)
  for (int idx = t; idx < kvlen * 8; idx += 256) {
    const int row = idx >> 3;
    const int c   = (idx & 7) << 3;
    async_b128(ks0 + (unsigned)(row * 72 + c) * 2,
               Kp + (size_t)(krow0 + row) * D_MODEL + col0 + c);
  }
  // ---- stage V transposed: Vs[d][key] (manual 2B scatter)
  for (int idx = t; idx < kvlen * 64; idx += 256) {
    const int r  = idx >> 6;
    const int dd = idx & 63;
    Vs[dd * vstride + r] = Vp[(size_t)(krow0 + r) * D_MODEL + col0 + dd];
  }
  wait_async0();
  __syncthreads();

  const int wave = t >> 5, lane = t & 31;
  const int lg = lane >> 4, ln = lane & 15;
  ushort_t* Pw = Ps + wave * (2 * 16 * 40);
  const int qb = wave * 32;

  BF16Frag qf[2][2];
#pragma unroll
  for (int qt = 0; qt < 2; ++qt) {
    const int row = qb + qt * 16 + ln;
#pragma unroll
    for (int ks = 0; ks < 2; ++ks) {
      const int kk = ks * 32 + lg * 8;
      qf[qt][ks].q[0] = *(const u32x4*)(&Qs[row * 72 + kk]);
      qf[qt][ks].q[1] = *(const u32x4*)(&Qs[row * 72 + kk + 16]);
    }
  }

  v8f oacc[2][4];
  float mrun[2][8], lrun[2][8];
#pragma unroll
  for (int qt = 0; qt < 2; ++qt) {
#pragma unroll
    for (int dt = 0; dt < 4; ++dt) oacc[qt][dt] = vzero8();
#pragma unroll
    for (int v = 0; v < 8; ++v) { mrun[qt][v] = -1e30f; lrun[qt][v] = 0.f; }
  }
  const float scale = 0.125f;

  for (int kb = 0; kb < kvlen; kb += 32) {
    BF16Frag kf[2][2];
#pragma unroll
    for (int nt = 0; nt < 2; ++nt) {
      const int key = kb + nt * 16 + ln;
#pragma unroll
      for (int ks = 0; ks < 2; ++ks) {
        const int d0 = ks * 32 + lg * 8;
        kf[nt][ks].q[0] = *(const u32x4*)(&Ks[key * 72 + d0]);
        kf[nt][ks].q[1] = *(const u32x4*)(&Ks[key * 72 + d0 + 16]);
      }
    }
    v8f sc[2][2];
#pragma unroll
    for (int qt = 0; qt < 2; ++qt)
#pragma unroll
      for (int nt = 0; nt < 2; ++nt) {
        v8f a = wmma_bf16(qf[qt][0], kf[nt][0], vzero8());
        sc[qt][nt] = wmma_bf16(qf[qt][1], kf[nt][1], a);
      }

#pragma unroll
    for (int qt = 0; qt < 2; ++qt) {
#pragma unroll
      for (int v = 0; v < 8; ++v) {
        const int qi = qb + qt * 16 + v + 8 * lg;
        float best = -1e30f;
#pragma unroll
        for (int nt = 0; nt < 2; ++nt) {
          float sv = sc[qt][nt][v] * scale;
          const int ki = kb + nt * 16 + ln;
          if (causal && (ki > qi)) sv = -1e30f;
          sc[qt][nt][v] = sv;
          best = fmaxf(best, sv);
        }
#pragma unroll
        for (int dsh = 1; dsh < 16; dsh <<= 1)
          best = fmaxf(best, __shfl_xor(best, dsh, 32));
        const float mold = mrun[qt][v];
        const float mnew = fmaxf(mold, best);
        const float corr = __expf(mold - mnew);
        mrun[qt][v] = mnew;
        float rsum = 0.f;
#pragma unroll
        for (int nt = 0; nt < 2; ++nt) {
          const float p = __expf(sc[qt][nt][v] - mnew);
          sc[qt][nt][v] = p;
          rsum += p;
        }
#pragma unroll
        for (int dsh = 1; dsh < 16; dsh <<= 1)
          rsum += __shfl_xor(rsum, dsh, 32);
        lrun[qt][v] = lrun[qt][v] * corr + rsum;
#pragma unroll
        for (int dt = 0; dt < 4; ++dt) oacc[qt][dt][v] *= corr;
      }
#pragma unroll
      for (int v = 0; v < 8; ++v) {
        const int m = v + 8 * lg;
#pragma unroll
        for (int nt = 0; nt < 2; ++nt)
          Pw[qt * 640 + m * 40 + nt * 16 + ln] =
              (ushort_t)f2bfu(sc[qt][nt][v]);
      }
    }

    BF16Frag pf[2];
#pragma unroll
    for (int qt = 0; qt < 2; ++qt) {
      pf[qt].q[0] = *(const u32x4*)(&Pw[qt * 640 + ln * 40 + lg * 8]);
      pf[qt].q[1] = *(const u32x4*)(&Pw[qt * 640 + ln * 40 + lg * 8 + 16]);
    }
#pragma unroll
    for (int dt = 0; dt < 4; ++dt) {
      BF16Frag vf;
      const int dd = dt * 16 + ln;
      vf.q[0] = *(const u32x4*)(&Vs[dd * vstride + kb + lg * 8]);
      vf.q[1] = *(const u32x4*)(&Vs[dd * vstride + kb + lg * 8 + 16]);
#pragma unroll
      for (int qt = 0; qt < 2; ++qt)
        oacc[qt][dt] = wmma_bf16(pf[qt], vf, oacc[qt][dt]);
    }
  }

#pragma unroll
  for (int qt = 0; qt < 2; ++qt) {
    float inv[8];
#pragma unroll
    for (int v = 0; v < 8; ++v) inv[v] = 1.f / lrun[qt][v];
#pragma unroll
    for (int dt = 0; dt < 4; ++dt) {
#pragma unroll
      for (int v = 0; v < 8; ++v) {
        const int m = qb + qt * 16 + v + 8 * lg;
        const int n = dt * 16 + ln;
        Ob[(size_t)(qrow0 + m) * D_MODEL + col0 + n] =
            (ushort_t)f2bfu(oacc[qt][dt][v] * inv[v]);
      }
    }
  }
}

// ---------------------------------------------------------------------------
// Host orchestration
// ---------------------------------------------------------------------------
static inline void launch_gemm(hipStream_t s, const ushort_t* A, const ushort_t* Bt,
                               const float* bias, const float* Res,
                               float* Cf, ushort_t* Cb,
                               int M, int N, int K, int lda, int ldb, int ldc,
                               int gelu, int addres, int skip) {
  dim3 grid((N + BN - 1) / BN, M / BM);
  gemm_bf16<<<grid, 256, 0, s>>>(A, Bt, bias, Res, Cf, Cb, N, K, lda, ldb, ldc,
                                 gelu, addres, skip);
}

extern "C" void kernel_launch(void* const* d_in, const int* in_sizes, int n_in,
                              void* d_out, int out_size, void* d_ws, size_t ws_size,
                              hipStream_t stream) {
  (void)in_sizes; (void)n_in; (void)out_size; (void)ws_size;
  const int*   ids   = (const int*)  d_in[0];
  const float* tok   = (const float*)d_in[1];
  const float* pos   = (const float*)d_in[2];
  const float* lnf_g = (const float*)d_in[3];
  const float* lnf_b = (const float*)d_in[4];
  const float* wq  = (const float*)d_in[5];  const float* bq  = (const float*)d_in[6];
  const float* cwq = (const float*)d_in[7];  const float* cbq = (const float*)d_in[8];
  const float* wk  = (const float*)d_in[9];  const float* bk  = (const float*)d_in[10];
  const float* cwk = (const float*)d_in[11]; const float* cbk = (const float*)d_in[12];
  const float* wv  = (const float*)d_in[13]; const float* bv  = (const float*)d_in[14];
  const float* cwv = (const float*)d_in[15]; const float* cbv = (const float*)d_in[16];
  const float* wo  = (const float*)d_in[17]; const float* bo  = (const float*)d_in[18];
  const float* cwo = (const float*)d_in[19]; const float* cbo = (const float*)d_in[20];
  const float* w1  = (const float*)d_in[21]; const float* b1  = (const float*)d_in[22];
  const float* w2  = (const float*)d_in[23]; const float* b2  = (const float*)d_in[24];
  const float* ln1g = (const float*)d_in[25]; const float* ln1b = (const float*)d_in[26];
  const float* lncg = (const float*)d_in[27]; const float* lncb = (const float*)d_in[28];
  const float* ln2g = (const float*)d_in[29]; const float* ln2b = (const float*)d_in[30];

  float* logits = (float*)d_out;

  const size_t R  = (size_t)NBATCH * SEQ;       // 4096
  const size_t RC = (size_t)NBATCH * (SEQ / 4); // 1024
  const size_t DD = (size_t)D_MODEL * D_MODEL;
  const size_t DF = (size_t)D_MODEL * DFF;

  // --- workspace carve (256B aligned) ---
  char* p = (char*)d_ws;
  auto alloc = [&](size_t bytes) -> char* {
    char* r = p;
    p += (bytes + 255) & ~(size_t)255;
    return r;
  };
  float*    x     = (float*)   alloc(R * D_MODEL * 4);   // residual stream
  ushort_t* h     = (ushort_t*)alloc(R * D_MODEL * 2);   // LN outputs
  ushort_t* qb_   = (ushort_t*)alloc(R * D_MODEL * 2);
  ushort_t* kb_   = (ushort_t*)alloc(R * D_MODEL * 2);
  ushort_t* vb_   = (ushort_t*)alloc(R * D_MODEL * 2);
  ushort_t* aob   = (ushort_t*)alloc(R * D_MODEL * 2);
  ushort_t* m1b   = (ushort_t*)alloc(R * DFF * 2);
  ushort_t* compb = (ushort_t*)alloc(RC * D_MODEL * 2);
  ushort_t* kcb   = (ushort_t*)alloc(RC * D_MODEL * 2);
  ushort_t* vcb   = (ushort_t*)alloc(RC * D_MODEL * 2);
  // pre-transposed bf16 weights  Wt[N][K]
  ushort_t* wqt  = (ushort_t*)alloc(NLAYER * DD * 2);
  ushort_t* wkt  = (ushort_t*)alloc(NLAYER * DD * 2);
  ushort_t* wvt  = (ushort_t*)alloc(NLAYER * DD * 2);
  ushort_t* wot  = (ushort_t*)alloc(NLAYER * DD * 2);
  ushort_t* cwqt = (ushort_t*)alloc(NLAYER * DD * 2);
  ushort_t* cwkt = (ushort_t*)alloc(NLAYER * DD * 2);
  ushort_t* cwvt = (ushort_t*)alloc(NLAYER * DD * 2);
  ushort_t* cwot = (ushort_t*)alloc(NLAYER * DD * 2);
  ushort_t* w1t  = (ushort_t*)alloc(NLAYER * DF * 2);    // [4096][1024]
  ushort_t* w2t  = (ushort_t*)alloc(NLAYER * DF * 2);    // [1024][4096]
  ushort_t* tokb = (ushort_t*)alloc((size_t)NVOCAB * D_MODEL * 2);
  (void)alloc(256 * 1024);  // pad: harmless OOB B-panel reads in ragged N tail

  // --- weight conversion (+transpose) ---
  {
    dim3 blk(256);
    dim3 gdd(D_MODEL / 32, D_MODEL / 32, NLAYER);
    cvt_t_kernel<<<gdd, blk, 0, stream>>>(wq,  wqt,  D_MODEL, D_MODEL);
    cvt_t_kernel<<<gdd, blk, 0, stream>>>(wk,  wkt,  D_MODEL, D_MODEL);
    cvt_t_kernel<<<gdd, blk, 0, stream>>>(wv,  wvt,  D_MODEL, D_MODEL);
    cvt_t_kernel<<<gdd, blk, 0, stream>>>(wo,  wot,  D_MODEL, D_MODEL);
    cvt_t_kernel<<<gdd, blk, 0, stream>>>(cwq, cwqt, D_MODEL, D_MODEL);
    cvt_t_kernel<<<gdd, blk, 0, stream>>>(cwk, cwkt, D_MODEL, D_MODEL);
    cvt_t_kernel<<<gdd, blk, 0, stream>>>(cwv, cwvt, D_MODEL, D_MODEL);
    cvt_t_kernel<<<gdd, blk, 0, stream>>>(cwo, cwot, D_MODEL, D_MODEL);
    cvt_t_kernel<<<dim3(DFF / 32, D_MODEL / 32, NLAYER), blk, 0, stream>>>(
        w1, w1t, D_MODEL, DFF);
    cvt_t_kernel<<<dim3(D_MODEL / 32, DFF / 32, NLAYER), blk, 0, stream>>>(
        w2, w2t, DFF, D_MODEL);
    const int n4 = (int)((size_t)NVOCAB * D_MODEL / 4);
    cvt_kernel<<<(n4 + 255) / 256, 256, 0, stream>>>(tok, tokb, n4);
  }

  // --- attention LDS sizes ---
  const int kvcap_self  = WIN;
  const int kvcap_cross = (NWIN - 1) * 64;
  const size_t smem_self =
      (size_t)(256 * 72 + kvcap_self * 72 + 64 * (kvcap_self + 8) + 8 * 2 * 16 * 40) * 2;
  const size_t smem_cross =
      (size_t)(256 * 72 + kvcap_cross * 72 + 64 * (kvcap_cross + 8) + 8 * 2 * 16 * 40) * 2;

  embed_kernel<<<dim3((unsigned)R), 256, 0, stream>>>(ids, tok, pos, x);
  compress_kernel<<<dim3((unsigned)RC), 256, 0, stream>>>(x, compb);

  for (int l = 0; l < NLAYER; ++l) {
    // ---------- self attention ----------
    ln_kernel<<<dim3((unsigned)R), 256, 0, stream>>>(x, ln1g + l * D_MODEL,
                                                     ln1b + l * D_MODEL, h);
    launch_gemm(stream, h, wqt + l * DD, bq + l * D_MODEL, nullptr, nullptr, qb_,
                (int)R, D_MODEL, D_MODEL, D_MODEL, D_MODEL, D_MODEL, 0, 0, 0);
    launch_gemm(stream, h, wkt + l * DD, bk + l * D_MODEL, nullptr, nullptr, kb_,
                (int)R, D_MODEL, D_MODEL, D_MODEL, D_MODEL, D_MODEL, 0, 0, 0);
    launch_gemm(stream, h, wvt + l * DD, bv + l * D_MODEL, nullptr, nullptr, vb_,
                (int)R, D_MODEL, D_MODEL, D_MODEL, D_MODEL, D_MODEL, 0, 0, 0);
    attn_kernel<<<dim3(NWIN, NHEAD, NBATCH), 256, smem_self, stream>>>(
        qb_, kb_, vb_, aob, SEQ, SEQ, WIN, WIN, 0, 0, 1, kvcap_self);
    launch_gemm(stream, aob, wot + l * DD, bo + l * D_MODEL, x, x, nullptr,
                (int)R, D_MODEL, D_MODEL, D_MODEL, D_MODEL, D_MODEL, 0, 1, 0);

    // ---------- cross attention to compressed past ----------
    ln_kernel<<<dim3((unsigned)R), 256, 0, stream>>>(x, lncg + l * D_MODEL,
                                                     lncb + l * D_MODEL, h);
    launch_gemm(stream, h, cwqt + l * DD, cbq + l * D_MODEL, nullptr, nullptr, qb_,
                (int)R, D_MODEL, D_MODEL, D_MODEL, D_MODEL, D_MODEL, 0, 0, 0);
    launch_gemm(stream, compb, cwkt + l * DD, cbk + l * D_MODEL, nullptr, nullptr, kcb,
                (int)RC, D_MODEL, D_MODEL, D_MODEL, D_MODEL, D_MODEL, 0, 0, 0);
    launch_gemm(stream, compb, cwvt + l * DD, cbv + l * D_MODEL, nullptr, nullptr, vcb,
                (int)RC, D_MODEL, D_MODEL, D_MODEL, D_MODEL, D_MODEL, 0, 0, 0);
    attn_kernel<<<dim3(NWIN - 1, NHEAD, NBATCH), 256, smem_cross, stream>>>(
        qb_, kcb, vcb, aob, SEQ, SEQ / 4, 0, 0, 64, 1, 0, kvcap_cross);
    launch_gemm(stream, aob, cwot + l * DD, cbo + l * D_MODEL, x, x, nullptr,
                (int)R, D_MODEL, D_MODEL, D_MODEL, D_MODEL, D_MODEL, 0, 1, 1);

    // ---------- MLP ----------
    ln_kernel<<<dim3((unsigned)R), 256, 0, stream>>>(x, ln2g + l * D_MODEL,
                                                     ln2b + l * D_MODEL, h);
    launch_gemm(stream, h, w1t + l * DF, b1 + l * DFF, nullptr, nullptr, m1b,
                (int)R, DFF, D_MODEL, D_MODEL, D_MODEL, DFF, 1, 0, 0);
    launch_gemm(stream, m1b, w2t + l * DF, b2 + l * D_MODEL, x, x, nullptr,
                (int)R, D_MODEL, DFF, DFF, DFF, D_MODEL, 0, 1, 0);
  }

  // ---------- final LN + logits ----------
  ln_kernel<<<dim3((unsigned)R), 256, 0, stream>>>(x, lnf_g, lnf_b, h);
  launch_gemm(stream, h, tokb, nullptr, nullptr, logits, nullptr,
              (int)R, NVOCAB, D_MODEL, D_MODEL, D_MODEL, NVOCAB, 0, 0, 0);
}
